// MM_DiTBlock_61203283968574
// MI455X (gfx1250) — matrix operand order repaired and checked by reading, not compiled, not run on hardware
//
#include <hip/hip_runtime.h>
#include <math.h>

// ---------------------------------------------------------------------------
// MM-DiT block for MI455X (gfx1250, wave32, WMMA).
// Compute-bound (~460 GFLOP vs ~300MB weights @ 23.3 TB/s): all GEMMs run on
// v_wmma_f32_16x16x32_bf16 (f32 accumulate). A/B tiles are staged to LDS with
// GLOBAL_LOAD_ASYNC_TO_LDS_B128 (double-buffered, ASYNCcnt-tracked) when the
// toolchain exposes the builtins; otherwise plain loads + LDS stores.
// Workspace need ~= 0.7 GB.
// ---------------------------------------------------------------------------

#define DEV __device__ __forceinline__

typedef __bf16 bf16_t;
typedef bf16_t v16bf __attribute__((ext_vector_type(16)));
typedef float  v8f   __attribute__((ext_vector_type(8)));

// async-to-LDS builtin prototype (from hipcc diagnostic): (int4 AS1*, int4 AS3*, Ii, Ii)
typedef int v4i __attribute__((ext_vector_type(4)));
typedef __attribute__((address_space(1))) v4i g_v4i;   // global
typedef __attribute__((address_space(3))) v4i l_v4i;   // LDS

#if __has_builtin(__builtin_amdgcn_global_load_async_to_lds_b128)
#define HAVE_ASYNC_LDS 1
#else
#define HAVE_ASYNC_LDS 0
#endif

DEV void wait_async_zero() {
#if HAVE_ASYNC_LDS
#if __has_builtin(__builtin_amdgcn_s_wait_asynccnt)
  __builtin_amdgcn_s_wait_asynccnt(0);
#else
  asm volatile("s_wait_asynccnt 0" ::: "memory");
#endif
#endif
}

constexpr int Dm    = 768;
constexpr int Hh    = 12;
constexpr int Ss    = 1024;
constexpr int Bb    = 2;
constexpr int Mrows = Bb * Ss;     // 2048 tokens
constexpr int E2    = 2 * Dm;      // 1536 concat head dim
constexpr int NOut  = Hh * E2;     // 18432
constexpr int MLPd  = 4 * Dm;      // 3072
constexpr int D6    = 6 * Dm;      // 4608
constexpr int NQKV  = Hh * Dm;     // 9216

DEV unsigned short f32_to_bf16(float f) {
  unsigned u = __float_as_uint(f);
  u += 0x7FFFu + ((u >> 16) & 1u);          // round-to-nearest-even
  return (unsigned short)(u >> 16);
}
DEV float bf16_to_f32(unsigned short h) {
  return __uint_as_float(((unsigned)h) << 16);
}
DEV float gelu_tanh(float x) {
  float x3 = x * x * x;
  return 0.5f * x * (1.f + tanhf(0.7978845608028654f * (x + 0.044715f * x3)));
}
DEV float wave_sum(float v) {
#pragma unroll
  for (int o = 16; o > 0; o >>= 1) v += __shfl_down(v, o, 32);
  return v;
}
DEV float wave_max(float v) {
#pragma unroll
  for (int o = 16; o > 0; o >>= 1) v = fmaxf(v, __shfl_down(v, o, 32));
  return v;
}

// --- WMMA fragment loads (layouts per CDNA5 ISA 7.12.2, wave32) -------------
// A (16x32, 16-bit): lane L -> row m=L%16; elements 0..7 hold K=(L/16)*8+0..7,
//                    elements 8..15 hold K=16+(L/16)*8+0..7.
DEV v16bf load_frag_a(const unsigned short* p) {
  union { uint4 q[2]; v16bf v; } u;
  u.q[0] = *reinterpret_cast<const uint4*>(p);        // k .. k+7
  u.q[1] = *reinterpret_cast<const uint4*>(p + 16);   // k+16 .. k+23
  return u.v;
}
// B (32x16, 16-bit), from B^T rows: lane L -> col n=L%16; elements 0..7 hold
// K = (L/16)*8 + 0..7, elements 8..15 hold K = 16+(L/16)*8 + 0..7.
DEV v16bf load_frag_b(const unsigned short* p) {
  union { uint4 q[2]; v16bf v; } u;
  u.q[0] = *reinterpret_cast<const uint4*>(p);
  u.q[1] = *reinterpret_cast<const uint4*>(p + 16);
  return u.v;
}

// --- generic batched WMMA GEMM ---------------------------------------------
struct GemmArgs {
  const unsigned short* A;  long long sAz; int lda;   // M x K, bf16 row-major
  const unsigned short* BT; long long sBz; int ldb;   // N x K, bf16 row-major
  float* C32;  unsigned short* C16;  long long sCz;  int ldc;
  const float* bias;                       // length-N (absolute column nn)
  const float* gate;  int gate_ld;         // EPI==2: out = resid + gate*gelu(acc)
  const float* resid; int resid_ld;
  int M, N, K;
  float alpha;
  int eoff;      // e offset for OUTMODE 1/2 (0 = x stream, Dm = c stream)
};

// OUTMODE: 0 linear, 1 q/k (b,h,s,e), 2 v-transposed (b,h,e,s), 3 o (b,s,h*E2+e)
// EPI:     0 none, 1 gelu, 2 resid + gate*gelu
template <int EPI, int OUTMODE, bool HAS_C32, bool HAS_C16>
__global__ void __launch_bounds__(256)
gemm_wmma_bf16(GemmArgs g) {
  __shared__ unsigned short sA[2][128 * 32];   // 16 KB
  __shared__ unsigned short sB[2][128 * 32];   // 16 KB

  const int z = blockIdx.z;
  const unsigned short* A  = g.A  + (long long)z * g.sAz;
  const unsigned short* BT = g.BT + (long long)z * g.sBz;
  const int tileM = blockIdx.y * 128;
  const int tileN = blockIdx.x * 128;
  const int tid  = threadIdx.x;
  const int lane = tid & 31;
  const int wave = tid >> 5;                 // 8 waves: 2 (M) x 4 (N)
  const int wm = (wave >> 2) * 64;
  const int wn = (wave & 3) * 32;
  const int lm   = lane & 15;
  const int kh8  = (lane >> 4) << 3;
  const int kh16 = (lane >> 4) << 4;

  // staging: 512 16B-chunks per matrix tile; thread moves rows r0 and r0+64
  const int r0  = tid >> 2;
  const int kc0 = (tid & 3) * 8;

  // stage tile [k0, k0+32) into buffer `buf`
  auto stage = [&](int buf, int k0) {
    const unsigned short* ga0 = A  + (long long)(tileM + r0)      * g.lda + k0 + kc0;
    const unsigned short* ga1 = A  + (long long)(tileM + r0 + 64) * g.lda + k0 + kc0;
    const unsigned short* gb0 = BT + (long long)(tileN + r0)      * g.ldb + k0 + kc0;
    const unsigned short* gb1 = BT + (long long)(tileN + r0 + 64) * g.ldb + k0 + kc0;
    unsigned short* la0 = &sA[buf][r0 * 32 + kc0];
    unsigned short* la1 = &sA[buf][(r0 + 64) * 32 + kc0];
    unsigned short* lb0 = &sB[buf][r0 * 32 + kc0];
    unsigned short* lb1 = &sB[buf][(r0 + 64) * 32 + kc0];
#if HAVE_ASYNC_LDS
    __builtin_amdgcn_global_load_async_to_lds_b128((g_v4i*)ga0, (l_v4i*)la0, 0, 0);
    __builtin_amdgcn_global_load_async_to_lds_b128((g_v4i*)ga1, (l_v4i*)la1, 0, 0);
    __builtin_amdgcn_global_load_async_to_lds_b128((g_v4i*)gb0, (l_v4i*)lb0, 0, 0);
    __builtin_amdgcn_global_load_async_to_lds_b128((g_v4i*)gb1, (l_v4i*)lb1, 0, 0);
#else
    *reinterpret_cast<uint4*>(la0) = *reinterpret_cast<const uint4*>(ga0);
    *reinterpret_cast<uint4*>(la1) = *reinterpret_cast<const uint4*>(ga1);
    *reinterpret_cast<uint4*>(lb0) = *reinterpret_cast<const uint4*>(gb0);
    *reinterpret_cast<uint4*>(lb1) = *reinterpret_cast<const uint4*>(gb1);
#endif
  };

  v8f acc[4][2] = {};

  const int tiles = g.K / 32;
  stage(0, 0);
  wait_async_zero();
  __syncthreads();

  for (int t = 0; t < tiles; ++t) {
    const int cur = t & 1;
    if (t + 1 < tiles) stage(cur ^ 1, (t + 1) * 32);   // async prefetch next tile

    v16bf af[4], bfx[2];
#pragma unroll
    for (int i = 0; i < 4; ++i)
      af[i] = load_frag_a(&sA[cur][(wm + i * 16 + lm) * 32 + kh8]);   // ds_load_b128 x2
#pragma unroll
    for (int j = 0; j < 2; ++j)
      bfx[j] = load_frag_b(&sB[cur][(wn + j * 16 + lm) * 32 + kh8]);
#pragma unroll
    for (int i = 0; i < 4; ++i)
#pragma unroll
      for (int j = 0; j < 2; ++j)
        acc[i][j] = __builtin_amdgcn_wmma_f32_16x16x32_bf16(
            false, af[i], false, bfx[j], (short)0, acc[i][j], false, false);

    wait_async_zero();     // per-wave ASYNCcnt drain, then block-wide barrier
    __syncthreads();
  }

  // C/D layout: n = lane%16, m = vgpr + 8*(lane/16)
  const int cn = lane & 15;
  const int rb = (lane >> 4) << 3;
  const int m0 = tileM + wm;
  const int n0 = tileN + wn;
#pragma unroll
  for (int i = 0; i < 4; ++i) {
#pragma unroll
    for (int j = 0; j < 2; ++j) {
      const int nn = n0 + j * 16 + cn;
      const float bia = g.bias ? g.bias[nn] : 0.f;
#pragma unroll
      for (int r = 0; r < 8; ++r) {
        const int mm = m0 + i * 16 + rb + r;
        float v = acc[i][j][r] * g.alpha + bia;
        if (EPI == 1) {
          v = gelu_tanh(v);
        } else if (EPI == 2) {
          v = g.resid[(long long)mm * g.resid_ld + nn] +
              g.gate [(long long)mm * g.gate_ld  + nn] * gelu_tanh(v);
        }
        long long idx;
        if (OUTMODE == 0) {
          idx = (long long)z * g.sCz + (long long)mm * g.ldc + nn;
        } else if (OUTMODE == 1) {            // q/k: (b, h, s, e)
          int h = nn / Dm, e = nn % Dm + g.eoff, b = mm / Ss, s = mm % Ss;
          idx = (((long long)(b * Hh + h)) * Ss + s) * E2 + e;
        } else if (OUTMODE == 2) {            // v: (b, h, e, s)
          int h = nn / Dm, e = nn % Dm + g.eoff, b = mm / Ss, s = mm % Ss;
          idx = (((long long)(b * Hh + h)) * E2 + e) * Ss + s;
        } else {                              // o: (b, s, h*E2 + e), z = b*Hh+h
          int b = z / Hh, h = z % Hh;
          idx = ((long long)b * Ss + mm) * (long long)NOut + (long long)h * E2 + nn;
        }
        if (HAS_C32) g.C32[idx] = v;
        if (HAS_C16) g.C16[idx] = f32_to_bf16(v);
      }
    }
  }
}

template <int EPI, int OUTMODE, bool HAS_C32, bool HAS_C16>
static void gemm_launch(const GemmArgs& g, int batch, hipStream_t stream) {
  dim3 grid((unsigned)(g.N / 128), (unsigned)(g.M / 128), (unsigned)batch);
  gemm_wmma_bf16<EPI, OUTMODE, HAS_C32, HAS_C16><<<grid, 256, 0, stream>>>(g);
}

// --- elementwise / reduction kernels ---------------------------------------
__global__ void silu_kernel(const float* __restrict__ in,
                            unsigned short* __restrict__ out, int n) {
  int t = blockIdx.x * 256 + threadIdx.x;
  if (t >= n) return;
  float v = in[t];
  out[t] = f32_to_bf16(v / (1.f + expf(-v)));
}

__global__ void transpose_cvt_kernel(const float* __restrict__ in,
                                     unsigned short* __restrict__ out,
                                     int K, int N) {   // in: K x N -> out: N x K
  long long t = (long long)blockIdx.x * 256 + threadIdx.x;
  if (t >= (long long)K * N) return;
  int n = (int)(t / K), k = (int)(t % K);
  out[t] = f32_to_bf16(in[(long long)k * N + n]);
}

__global__ void ln_mod_kernel(const float* __restrict__ in,
                              const float* __restrict__ mods,
                              int sh_off, int sc_off,
                              unsigned short* __restrict__ out) {
  __shared__ float red[16];
  const int row = blockIdx.x;
  const float* r = in + (long long)row * Dm;
  float vals[3], s = 0.f, s2 = 0.f;
#pragma unroll
  for (int i = 0; i < 3; ++i) {
    float v = r[threadIdx.x + 256 * i];
    vals[i] = v; s += v; s2 += v * v;
  }
  s = wave_sum(s); s2 = wave_sum(s2);
  const int lane = threadIdx.x & 31, w = threadIdx.x >> 5;
  if (lane == 0) { red[w] = s; red[8 + w] = s2; }
  __syncthreads();
  if (threadIdx.x == 0) {
    float a = 0.f, b = 0.f;
    for (int i = 0; i < 8; ++i) { a += red[i]; b += red[8 + i]; }
    red[0] = a; red[8] = b;
  }
  __syncthreads();
  const float mean = red[0] * (1.f / Dm);
  const float var  = red[8] * (1.f / Dm) - mean * mean;
  const float inv  = rsqrtf(var + 1e-6f);
  const float* mrow = mods + (long long)row * D6;
#pragma unroll
  for (int i = 0; i < 3; ++i) {
    int cidx = threadIdx.x + 256 * i;
    float o = (vals[i] - mean) * inv * (1.f + mrow[sc_off + cidx]) + mrow[sh_off + cidx];
    out[(long long)row * Dm + cidx] = f32_to_bf16(o);
  }
}

__global__ void softmax_kernel(const unsigned short* __restrict__ Sc,
                               unsigned short* __restrict__ At) {
  __shared__ float red[16];
  const long long row = blockIdx.x;
  const unsigned short* r = Sc + row * Ss;
  float v[4], mx = -3.4e38f;
#pragma unroll
  for (int i = 0; i < 4; ++i) {
    v[i] = bf16_to_f32(r[threadIdx.x * 4 + i]);
    mx = fmaxf(mx, v[i]);
  }
  mx = wave_max(mx);
  const int lane = threadIdx.x & 31, w = threadIdx.x >> 5;
  if (lane == 0) red[w] = mx;
  __syncthreads();
  if (threadIdx.x == 0) {
    float m = red[0];
    for (int i = 1; i < 8; ++i) m = fmaxf(m, red[i]);
    red[0] = m;
  }
  __syncthreads();
  mx = red[0];
  __syncthreads();
  float sum = 0.f;
#pragma unroll
  for (int i = 0; i < 4; ++i) { v[i] = expf(v[i] - mx); sum += v[i]; }
  sum = wave_sum(sum);
  if (lane == 0) red[w] = sum;
  __syncthreads();
  if (threadIdx.x == 0) {
    float a = 0.f;
    for (int i = 0; i < 8; ++i) a += red[i];
    red[0] = a;
  }
  __syncthreads();
  const float inv = 1.f / red[0];
#pragma unroll
  for (int i = 0; i < 4; ++i)
    At[row * Ss + threadIdx.x * 4 + i] = f32_to_bf16(v[i] * inv);
}

// ---------------------------------------------------------------------------
extern "C" void kernel_launch(void* const* d_in, const int* in_sizes, int n_in,
                              void* d_out, int out_size, void* d_ws, size_t ws_size,
                              hipStream_t stream) {
  (void)in_sizes; (void)n_in; (void)out_size; (void)ws_size;
  const float* x         = (const float*)d_in[0];
  const float* c         = (const float*)d_in[1];
  const float* y         = (const float*)d_in[2];
  const float* w_adaln_x = (const float*)d_in[3];
  const float* b_adaln_x = (const float*)d_in[4];
  const float* w_adaln_c = (const float*)d_in[5];
  const float* b_adaln_c = (const float*)d_in[6];
  const float* wq_x = (const float*)d_in[7];
  const float* wk_x = (const float*)d_in[8];
  const float* wv_x = (const float*)d_in[9];
  const float* wq_c = (const float*)d_in[10];
  const float* wk_c = (const float*)d_in[11];
  const float* wv_c = (const float*)d_in[12];
  const float* w_proj = (const float*)d_in[13];
  const float* b_proj = (const float*)d_in[14];
  const float* w_lin_x = (const float*)d_in[15];
  const float* b_lin_x = (const float*)d_in[16];
  const float* w_lin_c = (const float*)d_in[17];
  const float* b_lin_c = (const float*)d_in[18];
  const float* w_mlp1_x = (const float*)d_in[19];
  const float* b_mlp1_x = (const float*)d_in[20];
  const float* w_mlp2_x = (const float*)d_in[21];
  const float* b_mlp2_x = (const float*)d_in[22];
  const float* w_mlp1_c = (const float*)d_in[23];
  const float* b_mlp1_c = (const float*)d_in[24];
  const float* w_mlp2_c = (const float*)d_in[25];
  const float* b_mlp2_c = (const float*)d_in[26];
  float* out = (float*)d_out;

  // workspace bump allocator
  char* base = (char*)d_ws; size_t off = 0;
  auto allocB = [&](size_t bytes) -> void* {
    off = (off + 255) & ~(size_t)255;
    void* p = base + off; off += bytes; return p;
  };
  auto aU = [&](size_t n) { return (unsigned short*)allocB(n * 2); };
  auto aF = [&](size_t n) { return (float*)allocB(n * 4); };

  // bf16 transposed weights
  unsigned short* wt_adaln_x = aU((size_t)D6 * Dm);
  unsigned short* wt_adaln_c = aU((size_t)D6 * Dm);
  unsigned short* wtq_x = aU((size_t)NQKV * Dm);
  unsigned short* wtk_x = aU((size_t)NQKV * Dm);
  unsigned short* wtv_x = aU((size_t)NQKV * Dm);
  unsigned short* wtq_c = aU((size_t)NQKV * Dm);
  unsigned short* wtk_c = aU((size_t)NQKV * Dm);
  unsigned short* wtv_c = aU((size_t)NQKV * Dm);
  unsigned short* wt_proj  = aU((size_t)Dm * NOut);
  unsigned short* wt_lin_x = aU((size_t)Dm * Dm);
  unsigned short* wt_lin_c = aU((size_t)Dm * Dm);
  unsigned short* wt_mlp1_x = aU((size_t)MLPd * Dm);
  unsigned short* wt_mlp1_c = aU((size_t)MLPd * Dm);
  unsigned short* wt_mlp2_x = aU((size_t)Dm * MLPd);
  unsigned short* wt_mlp2_c = aU((size_t)Dm * MLPd);
  // activations
  unsigned short* sy16 = aU((size_t)Mrows * Dm);
  float* mods_x = aF((size_t)Mrows * D6);
  float* mods_c = aF((size_t)Mrows * D6);
  unsigned short* xm16 = aU((size_t)Mrows * Dm);
  unsigned short* cm16 = aU((size_t)Mrows * Dm);
  unsigned short* q16 = aU((size_t)Bb * Hh * Ss * E2);
  unsigned short* k16 = aU((size_t)Bb * Hh * Ss * E2);
  unsigned short* v16 = aU((size_t)Bb * Hh * Ss * E2);
  unsigned short* sc16 = aU((size_t)Bb * Hh * Ss * Ss);
  unsigned short* at16 = aU((size_t)Bb * Hh * Ss * Ss);
  unsigned short* o16  = aU((size_t)Mrows * NOut);
  unsigned short* oa16 = aU((size_t)Mrows * Dm);
  float* xattn = aF((size_t)Mrows * Dm);
  float* cattn = aF((size_t)Mrows * Dm);
  unsigned short* xm2 = aU((size_t)Mrows * Dm);
  unsigned short* cm2 = aU((size_t)Mrows * Dm);
  unsigned short* h1  = aU((size_t)Mrows * MLPd);

  auto tcvt = [&](const float* in, unsigned short* o, int K, int N) {
    long long tot = (long long)K * N;
    transpose_cvt_kernel<<<dim3((unsigned)((tot + 255) / 256)), 256, 0, stream>>>(in, o, K, N);
  };
  auto G = [&](const unsigned short* A, int lda, const unsigned short* BT, int ldb,
               int M, int N, int K) {
    GemmArgs g{}; g.A = A; g.lda = lda; g.BT = BT; g.ldb = ldb;
    g.M = M; g.N = N; g.K = K; g.alpha = 1.f; return g;
  };

  // 0) weight convert + transpose
  tcvt(w_adaln_x, wt_adaln_x, Dm, D6);
  tcvt(w_adaln_c, wt_adaln_c, Dm, D6);
  tcvt(wq_x, wtq_x, Dm, NQKV);  tcvt(wk_x, wtk_x, Dm, NQKV);  tcvt(wv_x, wtv_x, Dm, NQKV);
  tcvt(wq_c, wtq_c, Dm, NQKV);  tcvt(wk_c, wtk_c, Dm, NQKV);  tcvt(wv_c, wtv_c, Dm, NQKV);
  tcvt(w_proj, wt_proj, NOut, Dm);
  tcvt(w_lin_x, wt_lin_x, Dm, Dm);  tcvt(w_lin_c, wt_lin_c, Dm, Dm);
  tcvt(w_mlp1_x, wt_mlp1_x, Dm, MLPd);  tcvt(w_mlp1_c, wt_mlp1_c, Dm, MLPd);
  tcvt(w_mlp2_x, wt_mlp2_x, MLPd, Dm);  tcvt(w_mlp2_c, wt_mlp2_c, MLPd, Dm);

  // 1) silu(y)
  silu_kernel<<<dim3((Mrows * Dm + 255) / 256), 256, 0, stream>>>(y, sy16, Mrows * Dm);

  // 2) adaLN mods = silu(y) @ w_adaln + b   (f32 out)
  { GemmArgs g = G(sy16, Dm, wt_adaln_x, Dm, Mrows, D6, Dm);
    g.C32 = mods_x; g.ldc = D6; g.bias = b_adaln_x; gemm_launch<0,0,true,false>(g, 1, stream); }
  { GemmArgs g = G(sy16, Dm, wt_adaln_c, Dm, Mrows, D6, Dm);
    g.C32 = mods_c; g.ldc = D6; g.bias = b_adaln_c; gemm_launch<0,0,true,false>(g, 1, stream); }

  // 3) xm = mod(ln(x)), cm = mod(ln(c))
  ln_mod_kernel<<<Mrows, 256, 0, stream>>>(x, mods_x, 0, Dm, xm16);
  ln_mod_kernel<<<Mrows, 256, 0, stream>>>(c, mods_c, 0, Dm, cm16);

  // 4) q/k/v projections (q,k -> (b,h,s,e); v -> transposed (b,h,e,s))
  { GemmArgs g = G(xm16, Dm, wtq_x, Dm, Mrows, NQKV, Dm); g.C16 = q16; g.eoff = 0;  gemm_launch<0,1,false,true>(g, 1, stream); }
  { GemmArgs g = G(cm16, Dm, wtq_c, Dm, Mrows, NQKV, Dm); g.C16 = q16; g.eoff = Dm; gemm_launch<0,1,false,true>(g, 1, stream); }
  { GemmArgs g = G(xm16, Dm, wtk_x, Dm, Mrows, NQKV, Dm); g.C16 = k16; g.eoff = 0;  gemm_launch<0,1,false,true>(g, 1, stream); }
  { GemmArgs g = G(cm16, Dm, wtk_c, Dm, Mrows, NQKV, Dm); g.C16 = k16; g.eoff = Dm; gemm_launch<0,1,false,true>(g, 1, stream); }
  { GemmArgs g = G(xm16, Dm, wtv_x, Dm, Mrows, NQKV, Dm); g.C16 = v16; g.eoff = 0;  gemm_launch<0,2,false,true>(g, 1, stream); }
  { GemmArgs g = G(cm16, Dm, wtv_c, Dm, Mrows, NQKV, Dm); g.C16 = v16; g.eoff = Dm; gemm_launch<0,2,false,true>(g, 1, stream); }

  // 5) scores = q @ k^T / sqrt(768)   (batched over b*h = 24)
  { GemmArgs g = G(q16, E2, k16, E2, Ss, Ss, E2);
    g.sAz = (long long)Ss * E2; g.sBz = (long long)Ss * E2;
    g.C16 = sc16; g.sCz = (long long)Ss * Ss; g.ldc = Ss;
    g.alpha = (float)(1.0 / sqrt((double)Dm));
    gemm_launch<0,0,false,true>(g, Bb * Hh, stream); }

  // 6) softmax rows
  softmax_kernel<<<Bb * Hh * Ss, 256, 0, stream>>>(sc16, at16);

  // 7) o = attn @ v  -> scattered into (b, s, h*E2+e)
  { GemmArgs g = G(at16, Ss, v16, Ss, Ss, E2, Ss);
    g.sAz = (long long)Ss * Ss; g.sBz = (long long)E2 * Ss;
    g.C16 = o16; gemm_launch<0,3,false,true>(g, Bb * Hh, stream); }

  // 8) out_attn = o @ w_proj + b_proj
  { GemmArgs g = G(o16, NOut, wt_proj, NOut, Mrows, Dm, NOut);
    g.C16 = oa16; g.ldc = Dm; g.bias = b_proj; gemm_launch<0,0,false,true>(g, 1, stream); }

  // 9) x_attn = x + g_msa_x * gelu(out_attn @ w_lin_x + b);  same for c
  { GemmArgs g = G(oa16, Dm, wt_lin_x, Dm, Mrows, Dm, Dm);
    g.C32 = xattn; g.ldc = Dm; g.bias = b_lin_x;
    g.resid = x; g.resid_ld = Dm; g.gate = mods_x + 2 * Dm; g.gate_ld = D6;
    gemm_launch<2,0,true,false>(g, 1, stream); }
  { GemmArgs g = G(oa16, Dm, wt_lin_c, Dm, Mrows, Dm, Dm);
    g.C32 = cattn; g.ldc = Dm; g.bias = b_lin_c;
    g.resid = c; g.resid_ld = Dm; g.gate = mods_c + 2 * Dm; g.gate_ld = D6;
    gemm_launch<2,0,true,false>(g, 1, stream); }

  // 10) mlp-branch LN+mod
  ln_mod_kernel<<<Mrows, 256, 0, stream>>>(xattn, mods_x, 3 * Dm, 4 * Dm, xm2);
  ln_mod_kernel<<<Mrows, 256, 0, stream>>>(cattn, mods_c, 3 * Dm, 4 * Dm, cm2);

  // 11) x stream MLP: h1 = gelu(xm2 @ w1 + b1); x_out = x + g_mlp_x*gelu(h1 @ w2 + b2)
  { GemmArgs g = G(xm2, Dm, wt_mlp1_x, Dm, Mrows, MLPd, Dm);
    g.C16 = h1; g.ldc = MLPd; g.bias = b_mlp1_x; gemm_launch<1,0,false,true>(g, 1, stream); }
  { GemmArgs g = G(h1, MLPd, wt_mlp2_x, MLPd, Mrows, Dm, MLPd);
    g.C32 = out; g.ldc = Dm; g.bias = b_mlp2_x;
    g.resid = x; g.resid_ld = Dm; g.gate = mods_x + 5 * Dm; g.gate_ld = D6;
    gemm_launch<2,0,true,false>(g, 1, stream); }

  // 12) c stream MLP (reuses h1 after x-mlp2 completes in stream order)
  { GemmArgs g = G(cm2, Dm, wt_mlp1_c, Dm, Mrows, MLPd, Dm);
    g.C16 = h1; g.ldc = MLPd; g.bias = b_mlp1_c; gemm_launch<1,0,false,true>(g, 1, stream); }
  { GemmArgs g = G(h1, MLPd, wt_mlp2_c, MLPd, Mrows, Dm, MLPd);
    g.C32 = out + (size_t)Mrows * Dm; g.ldc = Dm; g.bias = b_mlp2_c;
    g.resid = c; g.resid_ld = Dm; g.gate = mods_c + 5 * Dm; g.gate_ld = D6;
    gemm_launch<2,0,true,false>(g, 1, stream); }
}
